// SparseResBlockC2S3d_858993459500
// MI455X (gfx1250) — matrix operand order repaired and checked by reading, not compiled, not run on hardware
//
#include <hip/hip_runtime.h>

// ---------------- problem constants (fixed by reference) ----------------
#define R1 32
#define R2 64
#define BAT 2
#define CCH 256          // channels C
#define COCH 256         // out channels CO
#define NPER 2048
#define NVOX (BAT * NPER)          // 4096 parent voxels
#define NCH  (NVOX * 8)            // 32768 child voxels
#define N1COLS (COCH * 8)          // 2048 (conv1 output cols)
#define N2COLS COCH                // 256
#define G1 (BAT * R1 * R1 * R1)    // 65536
#define G2 (BAT * R2 * R2 * R2)    // 524288
#define KT1 8                      // 256/32 K tiles
#define NT1 (N1COLS / 16)          // 128 N tiles for w1
#define NT2 (N2COLS / 16)          // 16 N tiles for w2
#define OUT_SUBDIV_OFF (NCH * COCH)   // 8,388,608 floats

// ---------------- WMMA vector types ----------------
typedef __attribute__((ext_vector_type(16))) __bf16        v16bf;
typedef __attribute__((ext_vector_type(8)))  float         v8f;
typedef __attribute__((ext_vector_type(4)))  unsigned int  v4u;
typedef __attribute__((ext_vector_type(4)))  int           i4;

struct Frag { union { v4u u4[2]; v16bf v; }; };

__device__ __forceinline__ unsigned short f2bf(float f) {
    unsigned u = __float_as_uint(f);
    unsigned r = u + 0x7FFFu + ((u >> 16) & 1u);   // round-to-nearest-even
    return (unsigned short)(r >> 16);
}
__device__ __forceinline__ float silu(float x) {
    return x / (1.0f + __expf(-x));
}

// ---------------- async global -> LDS copy (CDNA5 path, guarded) --------
#if defined(__gfx1250__) && __has_builtin(__builtin_amdgcn_global_load_async_to_lds_b128)
#define HAVE_ASYNC_LDS 1
#else
#define HAVE_ASYNC_LDS 0
#endif

__device__ __forceinline__ void async_copy16(const void* g, void* l) {
#if HAVE_ASYNC_LDS
    __builtin_amdgcn_global_load_async_to_lds_b128(
        (__attribute__((address_space(1))) i4*)(g),
        (__attribute__((address_space(3))) i4*)(l), 0, 0);
#else
    *(v4u*)l = *(const v4u*)g;
#endif
}
__device__ __forceinline__ void async_wait0() {
#if HAVE_ASYNC_LDS
#if __has_builtin(__builtin_amdgcn_s_wait_asynccnt)
    __builtin_amdgcn_s_wait_asynccnt(0);
#else
    asm volatile("s_wait_asynccnt 0" ::: "memory");
#endif
#endif
}

// ================= grid init =================
__global__ void k_init_grids(int* g1, int* g2) {
    int t = blockIdx.x * blockDim.x + threadIdx.x;
    if (t < G1) g1[t] = -1;
    if (t < G2) g2[t] = -1;
}

// ================= scatter rows into grids =================
__global__ void k_scatter(const int* __restrict__ coords, int* g1, int* g2) {
    int i = blockIdx.x * blockDim.x + threadIdx.x;
    if (i >= NVOX) return;
    int b = coords[4 * i + 0], x = coords[4 * i + 1];
    int y = coords[4 * i + 2], z = coords[4 * i + 3];
    g1[((b * R1 + x) * R1 + y) * R1 + z] = i;
    for (int j = 0; j < 8; ++j) {
        int cx = 2 * x + ((j >> 2) & 1);
        int cy = 2 * y + ((j >> 1) & 1);
        int cz = 2 * z + (j & 1);
        g2[((b * R2 + cx) * R2 + cy) * R2 + cz] = i * 8 + j;
    }
}

// ============ neighbor tables (invalid -> zero sentinel row) ============
__global__ void k_nbr1(const int* __restrict__ coords, const int* __restrict__ g1,
                       int* __restrict__ nbr) {
    int t = blockIdx.x * blockDim.x + threadIdx.x;
    if (t >= NVOX * 27) return;
    int i = t / 27, k = t % 27;
    int dx = k / 9 - 1, dy = (k / 3) % 3 - 1, dz = k % 3 - 1;
    int b = coords[4 * i + 0];
    int x = coords[4 * i + 1] + dx, y = coords[4 * i + 2] + dy, z = coords[4 * i + 3] + dz;
    int idx = -1;
    if (x >= 0 && x < R1 && y >= 0 && y < R1 && z >= 0 && z < R1)
        idx = g1[((b * R1 + x) * R1 + y) * R1 + z];
    nbr[t] = (idx < 0) ? NVOX : idx;          // NVOX = all-zero row of h
}

__global__ void k_nbr2(const int* __restrict__ coords, const int* __restrict__ g2,
                       int* __restrict__ nbr) {
    int t = blockIdx.x * blockDim.x + threadIdx.x;
    if (t >= NCH * 27) return;
    int r = t / 27, k = t % 27;
    int p = r >> 3, j = r & 7;
    int dx = k / 9 - 1, dy = (k / 3) % 3 - 1, dz = k % 3 - 1;
    int b = coords[4 * p + 0];
    int x = 2 * coords[4 * p + 1] + ((j >> 2) & 1) + dx;
    int y = 2 * coords[4 * p + 2] + ((j >> 1) & 1) + dy;
    int z = 2 * coords[4 * p + 3] + (j & 1) + dz;
    int idx = -1;
    if (x >= 0 && x < R2 && y >= 0 && y < R2 && z >= 0 && z < R2)
        idx = g2[((b * R2 + x) * R2 + y) * R2 + z];
    nbr[t] = (idx < 0) ? NCH : idx;           // NCH = all-zero row of hc
}

// ================= pack weights fp32 -> bf16 B-fragment layout ===========
// tile = 32(K) x 16(N).  lane l owns col n=l&15, element e holds K = (e>>3)*16 + (l>>4)*8 + (e&7),
// stored as 16 contiguous bf16 at tile*512 + lane*16 (ushort units).
__global__ void k_pack_w1(const float* __restrict__ w, unsigned short* __restrict__ p) {
    int t = blockIdx.x * blockDim.x + threadIdx.x;
    const int TOT = 27 * KT1 * NT1 * 512;
    if (t >= TOT) return;
    int e = t & 15, lane = (t >> 4) & 31, tile = t >> 9;
    int nt = tile & (NT1 - 1), kt = (tile >> 7) & 7, k = tile >> 10;
    int K = kt * 32 + ((e >> 3) << 4) + ((lane >> 4) << 3) + (e & 7);
    int n = nt * 16 + (lane & 15);
    p[t] = f2bf(w[(k * CCH + K) * N1COLS + n]);
}
__global__ void k_pack_w2(const float* __restrict__ w, unsigned short* __restrict__ p) {
    int t = blockIdx.x * blockDim.x + threadIdx.x;
    const int TOT = 27 * KT1 * NT2 * 512;
    if (t >= TOT) return;
    int e = t & 15, lane = (t >> 4) & 31, tile = t >> 9;
    int nt = tile & (NT2 - 1), kt = (tile >> 4) & 7, k = tile >> 7;
    int K = kt * 32 + ((e >> 3) << 4) + ((lane >> 4) << 3) + (e & 7);
    int n = nt * 16 + (lane & 15);
    p[t] = f2bf(w[(k * CCH + K) * N2COLS + n]);
}

// ================= subdiv logits =================
__global__ void k_subdiv(const float* __restrict__ feats, const float* __restrict__ w_sub,
                         const float* __restrict__ b_sub, float* __restrict__ subdiv) {
    int t = blockIdx.x * blockDim.x + threadIdx.x;
    if (t >= NVOX * 8) return;
    int p = t >> 3, j = t & 7;
    float acc = b_sub[j];
    const float* f = feats + p * CCH;
    for (int c = 0; c < CCH; ++c) acc += f[c] * w_sub[c * 8 + j];
    subdiv[t] = acc;
}

// ====== LN (affine) + SiLU -> bf16 (wave per row; row NVOX = zeros) =====
__global__ void k_ln1(const float* __restrict__ feats, const float* __restrict__ gamma,
                      const float* __restrict__ beta, unsigned short* __restrict__ h) {
    int row = blockIdx.x * 8 + (threadIdx.x >> 5);
    int lane = threadIdx.x & 31;
    if (row > NVOX) return;
    if (row == NVOX) {                        // zero sentinel row
        for (int i = 0; i < 8; ++i) h[row * CCH + lane + 32 * i] = 0;
        return;
    }
    const float* x = feats + row * CCH;
    float v[8], s = 0.f, s2 = 0.f;
    for (int i = 0; i < 8; ++i) {
        float f = x[lane + 32 * i];
        v[i] = f; s += f; s2 += f * f;
    }
    for (int m = 16; m >= 1; m >>= 1) { s += __shfl_xor(s, m, 32); s2 += __shfl_xor(s2, m, 32); }
    float mu = s * (1.f / CCH);
    float inv = rsqrtf(s2 * (1.f / CCH) - mu * mu + 1e-6f);
    for (int i = 0; i < 8; ++i) {
        int c = lane + 32 * i;
        float y = (v[i] - mu) * inv * gamma[c] + beta[c];
        h[row * CCH + c] = f2bf(silu(y));
    }
}

// === mask + LN (no affine) + SiLU -> bf16 (row NCH = zero sentinel) =====
__global__ void k_ln2(const float* __restrict__ out1, const float* __restrict__ subdiv,
                      unsigned short* __restrict__ hc) {
    int row = blockIdx.x * 8 + (threadIdx.x >> 5);
    int lane = threadIdx.x & 31;
    if (row > NCH) return;
    if (row == NCH || !(subdiv[row] > 0.f)) { // sentinel or dropped child
        for (int i = 0; i < 8; ++i) hc[row * COCH + lane + 32 * i] = 0;
        return;
    }
    const float* x = out1 + (size_t)row * COCH;
    float v[8], s = 0.f, s2 = 0.f;
    for (int i = 0; i < 8; ++i) {
        float f = x[lane + 32 * i];
        v[i] = f; s += f; s2 += f * f;
    }
    for (int m = 16; m >= 1; m >>= 1) { s += __shfl_xor(s, m, 32); s2 += __shfl_xor(s2, m, 32); }
    float mu = s * (1.f / COCH);
    float inv = rsqrtf(s2 * (1.f / COCH) - mu * mu + 1e-6f);
    for (int i = 0; i < 8; ++i)
        hc[row * COCH + lane + 32 * i] = f2bf(silu((v[i] - mu) * inv));
}

#define WMMA_BF16(A, B, C) \
    __builtin_amdgcn_wmma_f32_16x16x32_bf16(false, (A), false, (B), (short)0, (C), false, false)

// ===== GEMM1: out1[4096,2048] = sum_k gather(h) @ W1[k] + b1 =====
// block = 8 waves, one column group (64 cols) x 256 rows; each wave 32x64.
// B tiles staged per tap into LDS with async global->LDS copies.
__global__ void __launch_bounds__(256) k_gemm1(const unsigned short* __restrict__ h,
                                               const unsigned short* __restrict__ w1p,
                                               const int* __restrict__ nbr,
                                               const float* __restrict__ b1,
                                               float* __restrict__ out1) {
    __shared__ v4u ldsB[2048];                              // 32 KB: 8 kt x 4 tiles x 1KB
    const int COLGROUPS = N1COLS / 64;                      // 32
    int cg = blockIdx.x % COLGROUPS;
    int rowBlk = blockIdx.x / COLGROUPS;                    // 0..15
    int wave = threadIdx.x >> 5, lane = threadIdx.x & 31;
    int rowBase = rowBlk * 256 + wave * 32;
    int colBase = cg * 64;

    v8f acc[2][4];
    for (int m = 0; m < 2; ++m)
        for (int t = 0; t < 4; ++t)
            for (int i = 0; i < 8; ++i) acc[m][t][i] = 0.f;

    int r0 = rowBase + (lane & 15), r1 = r0 + 16;
    int klo = (lane >> 4) ? 8 : 0;
    char* lbase = (char*)ldsB;

    for (int k = 0; k < 27; ++k) {
        __syncthreads();                                    // LDS reuse fence
        for (int kt = 0; kt < KT1; ++kt) {
            const unsigned short* src =
                w1p + ((size_t)(k * KT1 + kt) * NT1 + cg * 4) * 512 + threadIdx.x * 8;
            async_copy16(src, lbase + kt * 4096 + threadIdx.x * 16);
        }
        async_wait0();
        __syncthreads();

        const unsigned short* a0 = h + (size_t)nbr[r0 * 27 + k] * CCH;
        const unsigned short* a1 = h + (size_t)nbr[r1 * 27 + k] * CCH;
        for (int kt = 0; kt < KT1; ++kt) {
            Frag A0, A1;
            A0.u4[0] = *(const v4u*)(a0 + kt * 32 + klo);
            A0.u4[1] = *(const v4u*)(a0 + kt * 32 + 16 + klo);
            A1.u4[0] = *(const v4u*)(a1 + kt * 32 + klo);
            A1.u4[1] = *(const v4u*)(a1 + kt * 32 + 16 + klo);
            for (int t = 0; t < 4; ++t) {
                const v4u* bp = (const v4u*)(lbase + kt * 4096 + t * 1024 + lane * 32);
                Frag B;
                B.u4[0] = bp[0]; B.u4[1] = bp[1];
                acc[0][t] = WMMA_BF16(A0.v, B.v, acc[0][t]);
                acc[1][t] = WMMA_BF16(A1.v, B.v, acc[1][t]);
            }
        }
    }
    int n = lane & 15, mb = (lane >> 4) * 8;
    for (int m = 0; m < 2; ++m)
        for (int t = 0; t < 4; ++t) {
            int col = colBase + t * 16 + n;
            float bias = b1[col];
            for (int v = 0; v < 8; ++v)
                out1[(size_t)(rowBase + m * 16 + mb + v) * N1COLS + col] = acc[m][t][v] + bias;
        }
}

// == GEMM2: d_out[32768,256] = (sum_k gather(hc)@W2[k] + b2 + skip)*mask ==
__global__ void __launch_bounds__(256) k_gemm2(const unsigned short* __restrict__ hc,
                                               const unsigned short* __restrict__ w2p,
                                               const int* __restrict__ nbr,
                                               const float* __restrict__ b2,
                                               const float* __restrict__ feats,
                                               const float* __restrict__ subdiv,
                                               float* __restrict__ out) {
    __shared__ v4u ldsB[2048];                              // 32 KB
    const int COLGROUPS = N2COLS / 64;                      // 4
    int cg = blockIdx.x % COLGROUPS;
    int rowBlk = blockIdx.x / COLGROUPS;                    // 0..127
    int wave = threadIdx.x >> 5, lane = threadIdx.x & 31;
    int rowBase = rowBlk * 256 + wave * 32;
    int colBase = cg * 64;

    v8f acc[2][4];
    for (int m = 0; m < 2; ++m)
        for (int t = 0; t < 4; ++t)
            for (int i = 0; i < 8; ++i) acc[m][t][i] = 0.f;

    int r0 = rowBase + (lane & 15), r1 = r0 + 16;
    int klo = (lane >> 4) ? 8 : 0;
    char* lbase = (char*)ldsB;

    for (int k = 0; k < 27; ++k) {
        __syncthreads();
        for (int kt = 0; kt < KT1; ++kt) {
            const unsigned short* src =
                w2p + ((size_t)(k * KT1 + kt) * NT2 + cg * 4) * 512 + threadIdx.x * 8;
            async_copy16(src, lbase + kt * 4096 + threadIdx.x * 16);
        }
        async_wait0();
        __syncthreads();

        const unsigned short* a0 = hc + (size_t)nbr[r0 * 27 + k] * COCH;
        const unsigned short* a1 = hc + (size_t)nbr[r1 * 27 + k] * COCH;
        for (int kt = 0; kt < KT1; ++kt) {
            Frag A0, A1;
            A0.u4[0] = *(const v4u*)(a0 + kt * 32 + klo);
            A0.u4[1] = *(const v4u*)(a0 + kt * 32 + 16 + klo);
            A1.u4[0] = *(const v4u*)(a1 + kt * 32 + klo);
            A1.u4[1] = *(const v4u*)(a1 + kt * 32 + 16 + klo);
            for (int t = 0; t < 4; ++t) {
                const v4u* bp = (const v4u*)(lbase + kt * 4096 + t * 1024 + lane * 32);
                Frag B;
                B.u4[0] = bp[0]; B.u4[1] = bp[1];
                acc[0][t] = WMMA_BF16(A0.v, B.v, acc[0][t]);
                acc[1][t] = WMMA_BF16(A1.v, B.v, acc[1][t]);
            }
        }
    }
    int n = lane & 15, mb = (lane >> 4) * 8;
    for (int m = 0; m < 2; ++m)
        for (int t = 0; t < 4; ++t) {
            int col = colBase + t * 16 + n;
            float bias = b2[col];
            for (int v = 0; v < 8; ++v) {
                int r = rowBase + m * 16 + mb + v;
                int p = r >> 3, j = r & 7;
                float mk = (subdiv[r] > 0.f) ? 1.f : 0.f;
                float skip = feats[p * CCH + j * 32 + (col >> 3)];
                out[(size_t)r * N2COLS + col] = (acc[m][t][v] + bias + skip) * mk;
            }
        }
}

// ========================= launch =========================
extern "C" void kernel_launch(void* const* d_in, const int* in_sizes, int n_in,
                              void* d_out, int out_size, void* d_ws, size_t ws_size,
                              hipStream_t stream) {
    const int*   coords = (const int*)d_in[0];
    const float* feats  = (const float*)d_in[1];
    const float* gamma  = (const float*)d_in[2];
    const float* beta   = (const float*)d_in[3];
    const float* w_sub  = (const float*)d_in[4];
    const float* b_sub  = (const float*)d_in[5];
    const float* w1     = (const float*)d_in[6];
    const float* b1     = (const float*)d_in[7];
    const float* w2     = (const float*)d_in[8];
    const float* b2     = (const float*)d_in[9];
    float* out = (float*)d_out;
    float* subdiv_out = out + OUT_SUBDIV_OFF;

    char* ws = (char*)d_ws;
    int*            grid1 = (int*)(ws + 0);                       // 256 KB
    int*            grid2 = (int*)(ws + 262144);                  // 2 MB
    int*            nbr1  = (int*)(ws + 2359296);                 // 432 KB
    int*            nbr2  = (int*)(ws + 2801664);                 // 3.4 MB
    unsigned short* h     = (unsigned short*)(ws + 6340608);      // (NVOX+1) rows bf16
    unsigned short* w1p   = (unsigned short*)(ws + 8438272);      // 27 MB
    unsigned short* hc    = (unsigned short*)(ws + 8438272);      // aliases w1p (dead after gemm1)
    unsigned short* w2p   = (unsigned short*)(ws + 36749824);     // 3.4 MB
    float*          out1  = (float*)(ws + 40288768);              // 32 MB (end ~70.4 MB)

    const int BS = 256;
    k_init_grids<<<(G2 + BS - 1) / BS, BS, 0, stream>>>(grid1, grid2);
    k_scatter<<<(NVOX + BS - 1) / BS, BS, 0, stream>>>(coords, grid1, grid2);
    k_nbr1<<<(NVOX * 27 + BS - 1) / BS, BS, 0, stream>>>(coords, grid1, nbr1);
    k_nbr2<<<(NCH * 27 + BS - 1) / BS, BS, 0, stream>>>(coords, grid2, nbr2);
    k_pack_w1<<<(27 * KT1 * NT1 * 512 + BS - 1) / BS, BS, 0, stream>>>(w1, w1p);
    k_pack_w2<<<(27 * KT1 * NT2 * 512 + BS - 1) / BS, BS, 0, stream>>>(w2, w2p);
    k_subdiv<<<(NVOX * 8 + BS - 1) / BS, BS, 0, stream>>>(feats, w_sub, b_sub, subdiv_out);
    k_ln1<<<(NVOX + 1 + 7) / 8, BS, 0, stream>>>(feats, gamma, beta, h);
    // GEMM1: 16 row blocks (256 rows) * 32 col groups = 512 blocks of 8 waves
    k_gemm1<<<(NVOX / 256) * (N1COLS / 64), BS, 0, stream>>>(h, w1p, nbr1, b1, out1);
    k_ln2<<<(NCH + 1 + 7) / 8, BS, 0, stream>>>(out1, subdiv_out, hc);
    // GEMM2: 128 row blocks * 4 col groups = 512 blocks of 8 waves
    k_gemm2<<<(NCH / 256) * (N2COLS / 64), BS, 0, stream>>>(hc, w2p, nbr2, b2, feats,
                                                           subdiv_out, out);
}
